// BAN_65807488909779
// MI455X (gfx1250) — matrix operand order repaired and checked by reading, not compiled, not run on hardware
//
#include <hip/hip_runtime.h>
#include <hip/hip_bf16.h>
#include <stdint.h>
#include <stddef.h>

// ---------------- problem constants ----------------
constexpr int kB    = 64;
constexpr int kNQ   = 32;
constexpr int kNK   = 256;
constexpr int kE    = 300;
constexpr int kEP   = 320;    // E padded to multiple of 32 for WMMA K-loop
constexpr int kH    = 1024;
constexpr int kG    = 8;
constexpr int kNOUT = 300;
constexpr int kNANS = 4000;
constexpr int kGH   = kG * kH;   // 8192

// ---------------- WMMA types ----------------
typedef __attribute__((ext_vector_type(16))) __bf16 v16bf;
typedef __attribute__((ext_vector_type(8)))  __bf16 v8bf;
typedef __attribute__((ext_vector_type(8)))  float  v8f;

// A-fragment: 16x32 bf16 tile, row-major source with leading dim ld.
// Lane L: row m = L&15, khalf = L>>4. Elements 0..7 -> K = kt + khalf*8 + e,
// elements 8..15 -> K = kt + 16 + khalf*8 + (e-8).   (ISA 7.12.2)
__device__ __forceinline__ v16bf load_A(const __bf16* A, int ld, int kt, int lane) {
    int m  = lane & 15;
    int kh = lane >> 4;
    const __bf16* p = A + (size_t)m * ld + kt + kh * 8;
    v8bf lo = *(const v8bf*)(p);
    v8bf hi = *(const v8bf*)(p + 16);
    return __builtin_shufflevector(lo, hi, 0,1,2,3,4,5,6,7,8,9,10,11,12,13,14,15);
}

// B-fragment: 32x16 bf16 tile from pre-transposed Bt (N x K, row-major, ld = K stride).
// Lane L: col n = L&15, kgroup = L>>4. Elements 0..7 -> K = kt + kgroup*8 + e,
// elements 8..15 -> K = kt + 16 + kgroup*8 + (e-8) (same as A).
__device__ __forceinline__ v16bf load_B(const __bf16* Bt, int ld, int kt, int lane) {
    int n  = lane & 15;
    int kg = lane >> 4;
    const __bf16* p = Bt + (size_t)n * ld + kt + kg * 8;
    v8bf lo = *(const v8bf*)(p);
    v8bf hi = *(const v8bf*)(p + 16);
    return __builtin_shufflevector(lo, hi, 0,1,2,3,4,5,6,7,8,9,10,11,12,13,14,15);
}

__device__ __forceinline__ v8f wmma_bf16(v16bf a, v16bf b, v8f c) {
    return __builtin_amdgcn_wmma_f32_16x16x32_bf16(false, a, false, b, (short)0, c, false, false);
}

// ---------------- kernel 1: gather + convert embeddings (pad E->320) ----------------
__global__ void k_embed(const int* __restrict__ he_q, const int* __restrict__ he_k,
                        const float* __restrict__ emb,
                        __bf16* __restrict__ embq, __bf16* __restrict__ embk) {
    int row = blockIdx.x;            // 0 .. B*NQ + B*NK - 1
    int k   = threadIdx.x;           // blockDim = 320
    int tok; __bf16* dst;
    if (row < kB * kNQ) { tok = he_q[row];            dst = embq + (size_t)row * kEP; }
    else                { int r = row - kB * kNQ; tok = he_k[r]; dst = embk + (size_t)r * kEP; }
    float v = (k < kE) ? emb[(size_t)tok * kE + k] : 0.0f;
    dst[k] = (__bf16)v;
}

// ---------------- kernel 2: transpose+convert Wq/Wk -> bf16 [H x EP] ----------------
__global__ void k_prep_w(const float* __restrict__ Wq, const float* __restrict__ Wk,
                         __bf16* __restrict__ WqT, __bf16* __restrict__ WkT) {
    int h = blockIdx.x;              // 0..H-1
    int e = threadIdx.x;             // blockDim = 320
    float vq = 0.0f, vk = 0.0f;
    if (e < kE) { vq = Wq[(size_t)e * kH + h]; vk = Wk[(size_t)e * kH + h]; }
    WqT[(size_t)h * kEP + e] = (__bf16)vq;
    WkT[(size_t)h * kEP + e] = (__bf16)vk;
}

// ---------------- kernel 3: transpose Wout -> f32 [NOUT x GH] ----------------
__global__ void k_prep_wout(const float* __restrict__ Wout, float* __restrict__ WoutT) {
    int n = blockIdx.y;                          // 0..299
    int k = blockIdx.x * 256 + threadIdx.x;      // 0..8191
    if (k < kGH) WoutT[(size_t)n * kGH + k] = Wout[(size_t)k * kNOUT + n];
}

// ---------------- kernel 4: projection GEMMs (hq / hk), WMMA bf16 ----------------
// C[M x H] = Aemb[M x EP] @ WT[H x EP]^T + bias. One wave -> 16x32 output (2 n-tiles
// sharing one A fragment: 2 WMMAs per A load).
template <bool IS_K>
__global__ void k_proj(const __bf16* __restrict__ Aemb,
                       const __bf16* __restrict__ WT,
                       const float* __restrict__ bias,
                       float* __restrict__ hq_f32,
                       __bf16* __restrict__ hk_bf16,
                       __bf16* __restrict__ hkT_bf16) {
    const int lane  = threadIdx.x & 31;
    const int wave  = threadIdx.x >> 5;                 // 0..7
    const int mbase = blockIdx.x * 16;
    const int nbase = blockIdx.y * 256 + wave * 32;     // 2 n-tiles per wave
    const __bf16* Arow  = Aemb + (size_t)mbase * kEP;
    const __bf16* Brow0 = WT   + (size_t)nbase * kEP;
    const __bf16* Brow1 = WT   + (size_t)(nbase + 16) * kEP;
    v8f acc0 = {}, acc1 = {};
    for (int kt = 0; kt < kEP; kt += 32) {
        if (kt + 32 < kEP) {
            __builtin_prefetch(Brow0 + (size_t)(lane & 15) * kEP + kt + 32, 0, 1);
            __builtin_prefetch(Brow1 + (size_t)(lane & 15) * kEP + kt + 32, 0, 1);
        }
        v16bf a  = load_A(Arow,  kEP, kt, lane);
        v16bf b0 = load_B(Brow0, kEP, kt, lane);
        v16bf b1 = load_B(Brow1, kEP, kt, lane);
        acc0 = wmma_bf16(a, b0, acc0);
        acc1 = wmma_bf16(a, b1, acc1);
    }
    const int mh = (lane >> 4) * 8;
    #pragma unroll
    for (int nt = 0; nt < 2; ++nt) {
        const int n  = nbase + nt * 16 + (lane & 15);
        const float bn = bias[n];
        const v8f& acc = nt ? acc1 : acc0;
        #pragma unroll
        for (int r = 0; r < 8; ++r) {
            int m   = mbase + mh + r;
            float v = acc[r] + bn;
            if (!IS_K) {
                hq_f32[(size_t)m * kH + n] = v;
            } else {
                __bf16 bv = (__bf16)v;
                hk_bf16[(size_t)m * kH + n] = bv;
                int b_ = m / kNK, kk = m % kNK;             // hkT[b][h][k]
                hkT_bf16[((size_t)b_ * kH + n) * kNK + kk] = bv;
            }
        }
    }
}

// ---------------- kernel 5a: Ag[b,g,q,d] = hq[b,q,d] * Watt[d,g]  (bf16) ----------------
__global__ void k_scale(const float* __restrict__ hq, const float* __restrict__ Watt,
                        __bf16* __restrict__ Ag) {
    const int bgq = blockIdx.x;                 // 0 .. B*G*NQ-1
    const int q   = bgq % kNQ;
    const int bg  = bgq / kNQ;
    const int b   = bg / kG, g = bg % kG;
    const float* hr = hq + ((size_t)b * kNQ + q) * kH;
    __bf16* dst = Ag + (size_t)bgq * kH;
    for (int d = threadIdx.x; d < kH; d += 256)
        dst[d] = (__bf16)(hr[d] * Watt[(size_t)d * kG + g]);
}

// ---------------- kernel 5b: logits[b,g] = Ag[b,g] @ hk^T + batt[g] ----------------
// Per block: one (b,g). 8 waves, each wave: mtiles {0,1} x ntiles {2w,2w+1}. K = H.
// hk_bf16 row-major [NK x H] *is* B^T for this GEMM.
__global__ void k_logits(const __bf16* __restrict__ Ag, const __bf16* __restrict__ hk_bf16,
                         const float* __restrict__ batt, float* __restrict__ logits) {
    const int bg = blockIdx.x, b = bg / kG, g = bg % kG;
    const int lane = threadIdx.x & 31;
    const int wave = threadIdx.x >> 5;
    const __bf16* Ab  = Ag      + (size_t)bg * kNQ * kH;
    const __bf16* Bb0 = hk_bf16 + ((size_t)b * kNK + (2 * wave)     * 16) * kH;
    const __bf16* Bb1 = hk_bf16 + ((size_t)b * kNK + (2 * wave + 1) * 16) * kH;

    v8f acc[2][2] = {};
    for (int kt = 0; kt < kH; kt += 32) {
        if (kt + 32 < kH) {
            __builtin_prefetch(Ab  + (size_t)(lane & 15) * kH + kt + 32, 0, 1);
            __builtin_prefetch(Bb0 + (size_t)(lane & 15) * kH + kt + 32, 0, 1);
            __builtin_prefetch(Bb1 + (size_t)(lane & 15) * kH + kt + 32, 0, 1);
        }
        v16bf a0 = load_A(Ab,           kH, kt, lane);
        v16bf a1 = load_A(Ab + 16 * kH, kH, kt, lane);
        v16bf b0 = load_B(Bb0, kH, kt, lane);
        v16bf b1 = load_B(Bb1, kH, kt, lane);
        acc[0][0] = wmma_bf16(a0, b0, acc[0][0]);
        acc[0][1] = wmma_bf16(a0, b1, acc[0][1]);
        acc[1][0] = wmma_bf16(a1, b0, acc[1][0]);
        acc[1][1] = wmma_bf16(a1, b1, acc[1][1]);
    }

    const float bb_g = batt[g];
    float* Lb = logits + (size_t)bg * kNQ * kNK;
    const int ncol = lane & 15, mh = (lane >> 4) * 8;
    #pragma unroll
    for (int mt = 0; mt < 2; ++mt)
        #pragma unroll
        for (int nt = 0; nt < 2; ++nt) {
            int nb = (2 * wave + nt) * 16 + ncol;
            #pragma unroll
            for (int r = 0; r < 8; ++r) {
                int q = mt * 16 + mh + r;
                Lb[(size_t)q * kNK + nb] = acc[mt][nt][r] + bb_g;
            }
        }
}

// ---------------- kernel 6: softmax over 8192 per (b,g); emit bf16 only ----------------
__global__ void k_softmax(const float* __restrict__ logits, __bf16* __restrict__ att_bf16) {
    const int bg = blockIdx.x;
    const float* x = logits + (size_t)bg * kNQ * kNK;
    __shared__ float red[256];
    const int tid = threadIdx.x;
    float mx = -3.4e38f;
    for (int i = tid; i < kNQ * kNK; i += 256) mx = fmaxf(mx, x[i]);
    red[tid] = mx; __syncthreads();
    for (int s = 128; s > 0; s >>= 1) { if (tid < s) red[tid] = fmaxf(red[tid], red[tid + s]); __syncthreads(); }
    mx = red[0]; __syncthreads();
    float sum = 0.0f;
    for (int i = tid; i < kNQ * kNK; i += 256) sum += __expf(x[i] - mx);
    red[tid] = sum; __syncthreads();
    for (int s = 128; s > 0; s >>= 1) { if (tid < s) red[tid] += red[tid + s]; __syncthreads(); }
    const float inv = 1.0f / red[0];
    __bf16* ab = att_bf16 + (size_t)bg * kNQ * kNK;
    for (int i = tid; i < kNQ * kNK; i += 256)
        ab[i] = (__bf16)(__expf(x[i] - mx) * inv);
}

// ---------------- kernel 7: pooled[b,g,d] = sum_q hq[b,q,d] * (att[b,g] @ hk[b])[q,d] ----------------
// T-tile via WMMA (att bf16 as A, hkT as B), fused q-reduction epilogue. Deterministic, no atomics.
__global__ void k_pool(const __bf16* __restrict__ att_bf16,
                       const __bf16* __restrict__ hkT_bf16,
                       const float* __restrict__ hq, float* __restrict__ pooled) {
    const int bg = blockIdx.x, b = bg / kG;
    const __bf16* Ab  = att_bf16 + (size_t)bg * kNQ * kNK;
    const __bf16* Bb  = hkT_bf16 + (size_t)b  * kH  * kNK;
    const float*  hqb = hq       + (size_t)b  * kNQ * kH;
    const int lane = threadIdx.x & 31;
    const int wave = threadIdx.x >> 5;

    for (int ntc = 0; ntc < 8; ++ntc) {
        const int nt = wave * 8 + ntc;       // 0..63  (d-tile)
        const int nb = nt * 16;
        const __bf16* Brow = Bb + (size_t)nb * kNK;
        v8f acc0 = {}, acc1 = {};
        for (int kt = 0; kt < kNK; kt += 32) {
            if (kt + 32 < kNK)
                __builtin_prefetch(Brow + (size_t)(lane & 15) * kNK + kt + 32, 0, 1);
            v16bf a0 = load_A(Ab,            kNK, kt, lane);   // q-rows 0..15
            v16bf a1 = load_A(Ab + 16 * kNK, kNK, kt, lane);   // q-rows 16..31
            v16bf bv = load_B(Brow, kNK, kt, lane);
            acc0 = wmma_bf16(a0, bv, acc0);
            acc1 = wmma_bf16(a1, bv, acc1);
        }
        const int d  = nb + (lane & 15);
        const int mh = (lane >> 4) * 8;
        float partial = 0.0f;
        #pragma unroll
        for (int r = 0; r < 8; ++r) {
            partial += acc0[r] * hqb[(size_t)(mh + r)      * kH + d];
            partial += acc1[r] * hqb[(size_t)(16 + mh + r) * kH + d];
        }
        partial += __shfl_down(partial, 16);          // merge M=+8 half-wave
        if (lane < 16) pooled[(size_t)bg * kH + d] = partial;
    }
}

// ---------------- kernel 8: out[b,n] = pooled[b,:] . WoutT[n,:] + bout[n] ----------------
__global__ void k_out(const float* __restrict__ pooled, const float* __restrict__ WoutT,
                      const float* __restrict__ bout, float* __restrict__ out) {
    const int b = blockIdx.x, n = blockIdx.y;
    const float* p = pooled + (size_t)b * kGH;
    const float* w = WoutT  + (size_t)n * kGH;
    __shared__ float red[256];
    float s = 0.0f;
    for (int i = threadIdx.x; i < kGH; i += 256) s += p[i] * w[i];
    red[threadIdx.x] = s; __syncthreads();
    for (int st = 128; st > 0; st >>= 1) { if (threadIdx.x < st) red[threadIdx.x] += red[threadIdx.x + st]; __syncthreads(); }
    if (threadIdx.x == 0) out[(size_t)b * kNOUT + n] = red[0] + bout[n];
}

// ---------------- kernel 9: sim[b,a] = out[b] . glove_cands[a]  (one wave per element) ----------------
__global__ void k_sim(const float* __restrict__ out, const float* __restrict__ cands,
                      float* __restrict__ sim) {
    const int gid  = blockIdx.x * 8 + (threadIdx.x >> 5);
    const int lane = threadIdx.x & 31;
    const int b = gid / kNANS, a = gid % kNANS;
    const float* o = out   + (size_t)b * kNOUT;
    const float* c = cands + (size_t)a * kNOUT;
    float s = 0.0f;
    for (int i = lane; i < kNOUT; i += 32) s += o[i] * c[i];
    #pragma unroll
    for (int off = 16; off > 0; off >>= 1) s += __shfl_down(s, off);
    if (lane == 0) sim[(size_t)b * kNANS + a] = s;
}

// ---------------- kernel 10: log_softmax over 4000 per b ----------------
__global__ void k_logsoftmax(const float* __restrict__ sim, float* __restrict__ outp) {
    const int b = blockIdx.x;
    const float* x = sim + (size_t)b * kNANS;
    __shared__ float red[256];
    const int tid = threadIdx.x;
    float mx = -3.4e38f;
    for (int i = tid; i < kNANS; i += 256) mx = fmaxf(mx, x[i]);
    red[tid] = mx; __syncthreads();
    for (int s = 128; s > 0; s >>= 1) { if (tid < s) red[tid] = fmaxf(red[tid], red[tid + s]); __syncthreads(); }
    mx = red[0]; __syncthreads();
    float sum = 0.0f;
    for (int i = tid; i < kNANS; i += 256) sum += expf(x[i] - mx);
    red[tid] = sum; __syncthreads();
    for (int s = 128; s > 0; s >>= 1) { if (tid < s) red[tid] += red[tid + s]; __syncthreads(); }
    const float lse = mx + logf(red[0]);
    for (int i = tid; i < kNANS; i += 256) outp[(size_t)b * kNANS + i] = x[i] - lse;
}

// ---------------- host launcher ----------------
extern "C" void kernel_launch(void* const* d_in, const int* in_sizes, int n_in,
                              void* d_out, int out_size, void* d_ws, size_t ws_size,
                              hipStream_t stream) {
    const int*   he_ques = (const int*)  d_in[0];
    const int*   he_kg   = (const int*)  d_in[1];
    const float* emb     = (const float*)d_in[2];
    const float* Wq      = (const float*)d_in[3];
    const float* bq      = (const float*)d_in[4];
    const float* Wk      = (const float*)d_in[5];
    const float* bk      = (const float*)d_in[6];
    const float* Watt    = (const float*)d_in[7];
    const float* batt    = (const float*)d_in[8];
    const float* Wout    = (const float*)d_in[9];
    const float* bout    = (const float*)d_in[10];
    const float* cands   = (const float*)d_in[11];

    char* base = (char*)d_ws;
    size_t off = 0;
    auto carve = [&](size_t bytes) -> char* {
        char* p = base + off;
        off = (off + bytes + 255) & ~(size_t)255;
        return p;
    };
    __bf16* embq_bf  = (__bf16*)carve((size_t)kB * kNQ * kEP * 2);
    __bf16* embk_bf  = (__bf16*)carve((size_t)kB * kNK * kEP * 2);
    __bf16* WqT      = (__bf16*)carve((size_t)kH * kEP * 2);
    __bf16* WkT      = (__bf16*)carve((size_t)kH * kEP * 2);
    float*  hq_f32   = (float*) carve((size_t)kB * kNQ * kH * 4);
    __bf16* hk_bf16  = (__bf16*)carve((size_t)kB * kNK * kH * 2);
    __bf16* hkT_bf16 = (__bf16*)carve((size_t)kB * kH * kNK * 2);
    __bf16* Ag_bf16  = (__bf16*)carve((size_t)kB * kG * kNQ * kH * 2);
    float*  logits   = (float*) carve((size_t)kB * kG * kNQ * kNK * 4);
    __bf16* att_bf16 = (__bf16*)carve((size_t)kB * kG * kNQ * kNK * 2);
    float*  pooled   = (float*) carve((size_t)kB * kG * kH * 4);
    float*  WoutT    = (float*) carve((size_t)kNOUT * kGH * 4);
    float*  outv     = (float*) carve((size_t)kB * kNOUT * 4);
    float*  sim      = (float*) carve((size_t)kB * kNANS * 4);

    // 1. gather embeddings -> bf16 (padded K)
    k_embed<<<kB * kNQ + kB * kNK, kEP, 0, stream>>>(he_ques, he_kg, emb, embq_bf, embk_bf);
    // 2-3. weight transposes
    k_prep_w<<<kH, kEP, 0, stream>>>(Wq, Wk, WqT, WkT);
    k_prep_wout<<<dim3(kGH / 256, kNOUT), 256, 0, stream>>>(Wout, WoutT);
    // 4. projections (WMMA)
    k_proj<false><<<dim3((kB * kNQ) / 16, kH / 256), 256, 0, stream>>>(embq_bf, WqT, bq, hq_f32, nullptr, nullptr);
    k_proj<true ><<<dim3((kB * kNK) / 16, kH / 256), 256, 0, stream>>>(embk_bf, WkT, bk, nullptr, hk_bf16, hkT_bf16);
    // 5. per-head scaled queries, then bilinear attention logits (WMMA)
    k_scale<<<kB * kG * kNQ, 256, 0, stream>>>(hq_f32, Watt, Ag_bf16);
    k_logits<<<kB * kG, 256, 0, stream>>>(Ag_bf16, hk_bf16, batt, logits);
    // 6. softmax over (q,k)
    k_softmax<<<kB * kG, 256, 0, stream>>>(logits, att_bf16);
    // 7. attention pooling (WMMA + fused q-reduction)
    k_pool<<<kB * kG, 256, 0, stream>>>(att_bf16, hkT_bf16, hq_f32, pooled);
    // 8. output projection
    k_out<<<dim3(kB, kNOUT), 256, 0, stream>>>(pooled, WoutT, bout, outv);
    // 9. similarity
    k_sim<<<(kB * kNANS) / 8, 256, 0, stream>>>(outv, cands, sim);
    // 10. log-softmax -> d_out
    k_logsoftmax<<<kB, 256, 0, stream>>>(sim, (float*)d_out);
}
